// xDeepFM_13030930776369
// MI455X (gfx1250) — matrix operand order repaired and checked by reading, not compiled, not run on hardware
//
#include <hip/hip_runtime.h>
#include <math.h>

// ---------------- problem constants ----------------
#define BATCH   8192
#define NF      26        // F
#define DD      32        // D
#define DS      33        // LDS row stride for X0 (bank-conflict pad)
#define H1P     132       // LDS row stride for transposed x1/x2 [d][h] (128+4)
#define NV      100000    // V
#define H1      128
#define H2      128
#define U1      400
#define U2      400
#define FD      (NF*DD)         // 832
#define K1TILES 22              // 676 -> 22*32
#define K2      3328            // 26*128 = 104*32 exact
#define K2TILES 104
#define LDA_H1  416             // 400 padded to 13*32
#define NB      4               // batch elements per CIN workgroup (A-frag reuse x8)

typedef __attribute__((ext_vector_type(16))) __bf16 v16bf;
typedef __attribute__((ext_vector_type(8)))  float  v8f;

// ---------------- WMMA fragment layout helpers (CDNA5 ISA 7.12.2) ----------
// A (16x32 bf16): lane L: M = L%16, hi = L/16.  element e of v16bf:
//   e=0..15 -> K = hi*16 + e (contiguous 16, same as B)
__device__ __forceinline__ int kmap_a(int e, int hi) {
  return hi * 16 + e;
}
// B (32x16 bf16): lane L: N = L%16, hi = L/16.  element e: K = hi*16 + e.

// Unguarded A-fragment load from a row-major bf16 row (row must cover padded K;
// zero-padded B-fragments make padded-K contributions vanish).
__device__ __forceinline__ v16bf load_a_rowmajor(const __bf16* row, int kbase, int hi) {
  union { v16bf v; float4 f[2]; } u;
  const int K0 = kbase + hi * 16;
  u.f[0] = *(const float4*)(row + K0);
  u.f[1] = *(const float4*)(row + K0 + 8);
  return u.v;
}

// ---------------- weight pre-pack kernels ----------------
// A-fragments for CIN: source W is [K][128] (ij-major, h fastest); A[M=h][K]=W[K][h].
__global__ void pack_cinA(const float* __restrict__ W, int Korig, int Ktiles,
                          __bf16* __restrict__ frag) {
  int gid = blockIdx.x * blockDim.x + threadIdx.x;
  int total = 8 * Ktiles * 32;
  if (gid >= total) return;
  int lane = gid & 31;
  int kt   = (gid >> 5) % Ktiles;
  int mt   = gid / (32 * Ktiles);
  int m    = mt * 16 + (lane & 15);
  int hi   = lane >> 4;
  __bf16* dst = frag + ((size_t)(mt * Ktiles + kt) * 32 + lane) * 16;
  #pragma unroll
  for (int e = 0; e < 16; ++e) {
    int K = kt * 32 + kmap_a(e, hi);
    float v = (K < Korig) ? W[(size_t)K * 128 + m] : 0.f;
    dst[e] = (__bf16)v;
  }
}

// B-fragments for DNN GEMMs: source W row-major [K][N].
__global__ void pack_B(const float* __restrict__ W, int Korig, int N,
                       int Ktiles, int Ntiles, __bf16* __restrict__ frag) {
  int gid = blockIdx.x * blockDim.x + threadIdx.x;
  int total = Ktiles * Ntiles * 32;
  if (gid >= total) return;
  int lane = gid & 31;
  int nt   = (gid >> 5) % Ntiles;
  int kt   = gid / (32 * Ntiles);
  int n    = nt * 16 + (lane & 15);
  int hi   = lane >> 4;
  __bf16* dst = frag + ((size_t)(kt * Ntiles + nt) * 32 + lane) * 16;
  #pragma unroll
  for (int e = 0; e < 16; ++e) {
    int K = kt * 32 + hi * 16 + e;
    float v = (K < Korig) ? W[(size_t)K * N + n] : 0.f;
    dst[e] = (__bf16)v;
  }
}

// ---------------- embedding gather (f32 table -> bf16 rows) ----------------
__global__ void gather_kernel(const int* __restrict__ sparse,
                              const float* __restrict__ E,
                              __bf16* __restrict__ embedB) {
  int t = blockIdx.x * blockDim.x + threadIdx.x;
  if (t >= BATCH * NF) return;
  int b = t / NF, f = t - b * NF;
  int idx = sparse[t];
  const float4* src = (const float4*)(E + ((size_t)f * NV + (size_t)idx) * DD);
  union { __bf16 e[32]; float4 f4[4]; } o;
  #pragma unroll
  for (int c = 0; c < 8; ++c) {
    float4 v = src[c];
    o.e[c * 4 + 0] = (__bf16)v.x;
    o.e[c * 4 + 1] = (__bf16)v.y;
    o.e[c * 4 + 2] = (__bf16)v.z;
    o.e[c * 4 + 3] = (__bf16)v.w;
  }
  float4* dst = (float4*)(embedB + (size_t)b * FD + f * DD);
  #pragma unroll
  for (int c = 0; c < 4; ++c) dst[c] = o.f4[c];
}

__global__ void linear_kernel(const int* __restrict__ sparse,
                              const float* __restrict__ w_lin,
                              const float* __restrict__ b_lin,
                              float* __restrict__ lin) {
  int b = blockIdx.x * blockDim.x + threadIdx.x;
  if (b >= BATCH) return;
  float s = b_lin[0];
  #pragma unroll
  for (int f = 0; f < NF; ++f) s += (float)sparse[b * NF + f] * w_lin[f];
  lin[b] = s;
}

// ---------------- CIN kernel: NB batch elements per workgroup ----------------
__global__ __launch_bounds__(256) void cin_kernel(const __bf16* __restrict__ embedB,
                                                  const __bf16* __restrict__ W0frag,
                                                  const __bf16* __restrict__ W1frag,
                                                  float* __restrict__ cin) {
  __shared__ float X0s[NB * NF * DS];    // per-b [i][d], stride 33
  __shared__ float x1t[NB * DD * H1P];   // per-b transposed [d][h], stride 132
  __shared__ float x2t[NB * DD * H1P];
  const int  t  = threadIdx.x;
  const long b0 = (long)blockIdx.x * NB;

  for (int k = t; k < NB * FD; k += 256) {
    int bi = k / FD, kk = k - bi * FD;
    int i = kk >> 5, d0 = kk & 31;
    X0s[bi * (NF * DS) + i * DS + d0] = (float)embedB[(b0 + bi) * FD + kk];
  }
  __syncthreads();

  const int w = t >> 5, lane = t & 31;
  const int l16 = lane & 15, hi = lane >> 4;
  const int mt = w;  // 8 waves -> 8 M-tiles of 16 rows (H1 = 128)
  const v8f vzero = {};

  // ---- layer 1: x1[h,d] = sum_ij W0[ij,h] * X0[i,d]*X0[j,d] ----
  {
    v8f acc[NB][2];
    #pragma unroll
    for (int bi = 0; bi < NB; ++bi) { acc[bi][0] = vzero; acc[bi][1] = vzero; }
    for (int kt = 0; kt < K1TILES; ++kt) {
      v16bf a = *(const v16bf*)(W0frag + ((size_t)(mt * K1TILES + kt) * 32 + lane) * 16);
      const int kb = kt * 32 + hi * 16;
      // per-kt index math, shared by all NB*2 fragments.  Clamp OOB K: the A
      // fragment is zero-padded there, so any finite B value contributes 0.
      int offI[16], offJ[16];
      #pragma unroll
      for (int e = 0; e < 16; ++e) {
        unsigned kk = kb + e;
        unsigned kc = (kk < 676u) ? kk : 0u;
        unsigned i = kc / 26u, j = kc - i * 26u;
        offI[e] = (int)(i * DS);
        offJ[e] = (int)(j * DS);
      }
      #pragma unroll
      for (int bi = 0; bi < NB; ++bi) {
        const float* X = X0s + bi * (NF * DS);
        #pragma unroll
        for (int nt = 0; nt < 2; ++nt) {
          const int d = nt * 16 + l16;
          union { v16bf v; __bf16 e[16]; } bu;
          #pragma unroll
          for (int e = 0; e < 16; ++e) {
            bu.e[e] = (__bf16)(X[offI[e] + d] * X[offJ[e] + d]);
          }
          acc[bi][nt] = __builtin_amdgcn_wmma_f32_16x16x32_bf16(
              false, a, false, bu.v, (short)0, acc[bi][nt], false, false);
        }
      }
    }
    // C frag: lane holds h = mt*16 + 8*hi + r (r contiguous) at column d
    #pragma unroll
    for (int bi = 0; bi < NB; ++bi) {
      #pragma unroll
      for (int nt = 0; nt < 2; ++nt) {
        const int d = nt * 16 + l16;
        float4* xw = (float4*)(x1t + bi * (DD * H1P) + d * H1P + mt * 16 + 8 * hi);
        float4 w0 = {acc[bi][nt][0], acc[bi][nt][1], acc[bi][nt][2], acc[bi][nt][3]};
        float4 w1 = {acc[bi][nt][4], acc[bi][nt][5], acc[bi][nt][6], acc[bi][nt][7]};
        xw[0] = w0; xw[1] = w1;     // 2x ds_store_b128
      }
    }
  }
  __syncthreads();

  // cin part 1: sum over d (NB*128 sums by 256 threads, uniform 2-iteration loop)
  for (int q = t; q < NB * H1; q += 256) {
    int bi = q >> 7, h = q & 127;
    float s = 0.f;
    #pragma unroll
    for (int d2 = 0; d2 < DD; ++d2) s += x1t[bi * (DD * H1P) + d2 * H1P + h];
    cin[(b0 + bi) * 256 + h] = s;
  }

  // ---- layer 2: x2[h,d] = sum_{i,j} W1c[i,j,h] * X0[i,d]*x1[j,d] ----
  {
    v8f acc[NB][2];
    #pragma unroll
    for (int bi = 0; bi < NB; ++bi) { acc[bi][0] = vzero; acc[bi][1] = vzero; }
    for (int kt = 0; kt < K2TILES; ++kt) {
      const __bf16* ap = W1frag + ((size_t)(mt * K2TILES + kt) * 32 + lane) * 16;
      if (kt + 1 < K2TILES) __builtin_prefetch(ap + 512, 0, 3);  // global_prefetch_b8
      v16bf a = *(const v16bf*)ap;
      const int kb = kt * 32 + hi * 16;  // multiple of 16 -> same i for all 16 elems
      const int i  = kb >> 7;
      const int jb = kb & 127;
      #pragma unroll
      for (int bi = 0; bi < NB; ++bi) {
        const float* X  = X0s + bi * (NF * DS);
        const float* XT = x1t + bi * (DD * H1P);
        #pragma unroll
        for (int nt = 0; nt < 2; ++nt) {
          const int d = nt * 16 + l16;
          const float xi = X[i * DS + d];
          const float4* xr = (const float4*)(XT + d * H1P + jb);  // 4x ds_load_b128
          float4 p0 = xr[0], p1 = xr[1], p2 = xr[2], p3 = xr[3];
          union { v16bf v; __bf16 e[16]; } bu;
          bu.e[0]  = (__bf16)(xi * p0.x); bu.e[1]  = (__bf16)(xi * p0.y);
          bu.e[2]  = (__bf16)(xi * p0.z); bu.e[3]  = (__bf16)(xi * p0.w);
          bu.e[4]  = (__bf16)(xi * p1.x); bu.e[5]  = (__bf16)(xi * p1.y);
          bu.e[6]  = (__bf16)(xi * p1.z); bu.e[7]  = (__bf16)(xi * p1.w);
          bu.e[8]  = (__bf16)(xi * p2.x); bu.e[9]  = (__bf16)(xi * p2.y);
          bu.e[10] = (__bf16)(xi * p2.z); bu.e[11] = (__bf16)(xi * p2.w);
          bu.e[12] = (__bf16)(xi * p3.x); bu.e[13] = (__bf16)(xi * p3.y);
          bu.e[14] = (__bf16)(xi * p3.z); bu.e[15] = (__bf16)(xi * p3.w);
          acc[bi][nt] = __builtin_amdgcn_wmma_f32_16x16x32_bf16(
              false, a, false, bu.v, (short)0, acc[bi][nt], false, false);
        }
      }
    }
    #pragma unroll
    for (int bi = 0; bi < NB; ++bi) {
      #pragma unroll
      for (int nt = 0; nt < 2; ++nt) {
        const int d = nt * 16 + l16;
        float4* xw = (float4*)(x2t + bi * (DD * H1P) + d * H1P + mt * 16 + 8 * hi);
        float4 w0 = {acc[bi][nt][0], acc[bi][nt][1], acc[bi][nt][2], acc[bi][nt][3]};
        float4 w1 = {acc[bi][nt][4], acc[bi][nt][5], acc[bi][nt][6], acc[bi][nt][7]};
        xw[0] = w0; xw[1] = w1;
      }
    }
  }
  __syncthreads();

  for (int q = t; q < NB * H2; q += 256) {
    int bi = q >> 7, h = q & 127;
    float s = 0.f;
    #pragma unroll
    for (int d2 = 0; d2 < DD; ++d2) s += x2t[bi * (DD * H1P) + d2 * H1P + h];
    cin[(b0 + bi) * 256 + H1 + h] = s;
  }
}

// ---------------- DNN GEMM: out = relu(A@W + bias), bf16 WMMA ----------------
// A rows must cover Ktiles*32 columns (zero/benign padding); Bfrag zero-padded in K.
__global__ __launch_bounds__(256) void gemm_bias_relu(
    const __bf16* __restrict__ A, int lda, int Ktiles,
    const __bf16* __restrict__ Bfrag, int Ntiles, int ntn,
    const float* __restrict__ bias,
    void* __restrict__ out, int ldo, int out_bf16, int padzero) {
  const int w = threadIdx.x >> 5, lane = threadIdx.x & 31;
  const int m16 = lane & 15, hi = lane >> 4;
  const long rowbase = (long)blockIdx.x * 128 + (long)w * 16;
  const __bf16* arow = A + (rowbase + m16) * (size_t)lda;
  const int nt0 = blockIdx.y * ntn;
  for (int nt = nt0; nt < nt0 + ntn; ++nt) {
    v8f acc = {};
    for (int kt = 0; kt < Ktiles; ++kt) {
      v16bf a  = load_a_rowmajor(arow, kt * 32, hi);
      v16bf bb = *(const v16bf*)(Bfrag + ((size_t)(kt * Ntiles + nt) * 32 + lane) * 16);
      acc = __builtin_amdgcn_wmma_f32_16x16x32_bf16(false, a, false, bb,
                                                    (short)0, acc, false, false);
    }
    const int n = nt * 16 + m16;
    const float bv = bias[n];
    #pragma unroll
    for (int r = 0; r < 8; ++r) {
      long row = rowbase + r + 8 * hi;
      float v = acc[r] + bv;
      v = (v > 0.f) ? v : 0.f;
      if (out_bf16) ((__bf16*)out)[(size_t)row * ldo + n] = (__bf16)v;
      else          ((float*)out)[(size_t)row * ldo + n] = v;
    }
  }
  // zero-fill pad columns [Ntiles*16, Ntiles*16+16) so the next GEMM can load unguarded
  if (padzero && blockIdx.y == gridDim.y - 1) {
    const int col = Ntiles * 16 + m16;
    #pragma unroll
    for (int r = 0; r < 8; ++r) {
      long row = rowbase + r + 8 * hi;
      ((__bf16*)out)[(size_t)row * ldo + col] = (__bf16)0.f;
    }
  }
}

// ---------------- final concat-dot + sigmoid ----------------
__global__ void final_kernel(const float* __restrict__ lin,
                             const float* __restrict__ cin,
                             const float* __restrict__ h2,
                             const float* __restrict__ dense,
                             const float* __restrict__ w_out,
                             const float* __restrict__ b_out,
                             float* __restrict__ out) {
  int b = blockIdx.x * blockDim.x + threadIdx.x;
  if (b >= BATCH) return;
  float s = b_out[0] + lin[b] * w_out[0];
  const float* c = cin + (size_t)b * 256;
  for (int h = 0; h < 256; ++h) s += c[h] * w_out[1 + h];
  const float* hh = h2 + (size_t)b * U2;
  for (int u = 0; u < U2; ++u) s += hh[u] * w_out[257 + u];
  const float* dd = dense + (size_t)b * 13;
  #pragma unroll
  for (int k = 0; k < 13; ++k) s += dd[k] * w_out[657 + k];
  out[b] = 1.f / (1.f + expf(-s));
}

// ---------------- host launcher ----------------
extern "C" void kernel_launch(void* const* d_in, const int* in_sizes, int n_in,
                              void* d_out, int out_size, void* d_ws, size_t ws_size,
                              hipStream_t stream) {
  const float* dense  = (const float*)d_in[0];
  const int*   sparse = (const int*)  d_in[1];
  const float* E      = (const float*)d_in[2];
  const float* Wc0    = (const float*)d_in[3];   // (26,26,128)  flat (i*26+j)*128+h
  const float* Wc1    = (const float*)d_in[4];   // (26,128,128) flat (i*128+j)*128+h
  const float* W1     = (const float*)d_in[5];   // (832,400)
  const float* b1     = (const float*)d_in[6];
  const float* W2     = (const float*)d_in[7];   // (400,400)
  const float* b2     = (const float*)d_in[8];
  const float* w_lin  = (const float*)d_in[9];
  const float* b_lin  = (const float*)d_in[10];
  const float* w_out  = (const float*)d_in[11];
  const float* b_out  = (const float*)d_in[12];
  float* out = (float*)d_out;

  char* ws = (char*)d_ws;
  size_t off = 0;
  auto take = [&](size_t bytes) -> char* {
    char* p = ws + off;
    off += (bytes + 255) & ~(size_t)255;
    return p;
  };
  __bf16* embedB  = (__bf16*)take((size_t)BATCH * FD * 2);                 // 13.6 MB
  __bf16* h1buf   = (__bf16*)take((size_t)BATCH * LDA_H1 * 2);             // 6.8 MB
  float*  h2buf   = (float*) take((size_t)BATCH * U2 * 4);                 // 13.1 MB
  float*  cinbuf  = (float*) take((size_t)BATCH * 256 * 4);                // 8.4 MB
  float*  linbuf  = (float*) take((size_t)BATCH * 4);
  __bf16* W0frag  = (__bf16*)take((size_t)8 * K1TILES * 32 * 16 * 2);      // 180 KB
  __bf16* W1cfrag = (__bf16*)take((size_t)8 * K2TILES * 32 * 16 * 2);      // 852 KB
  __bf16* W1frag  = (__bf16*)take((size_t)26 * 25 * 32 * 16 * 2);          // 666 KB
  __bf16* W2frag  = (__bf16*)take((size_t)13 * 25 * 32 * 16 * 2);          // 333 KB

  // weight pre-pack (tiny, amortized into the graph)
  pack_cinA<<<(8 * K1TILES * 32 + 255) / 256, 256, 0, stream>>>(Wc0, NF * NF, K1TILES, W0frag);
  pack_cinA<<<(8 * K2TILES * 32 + 255) / 256, 256, 0, stream>>>(Wc1, K2, K2TILES, W1cfrag);
  pack_B<<<(26 * 25 * 32 + 255) / 256, 256, 0, stream>>>(W1, FD, U1, 26, 25, W1frag);
  pack_B<<<(13 * 25 * 32 + 255) / 256, 256, 0, stream>>>(W2, U1, U2, 13, 25, W2frag);

  // embeddings + linear term
  gather_kernel<<<(BATCH * NF + 255) / 256, 256, 0, stream>>>(sparse, E, embedB);
  linear_kernel<<<BATCH / 256, 256, 0, stream>>>(sparse, w_lin, b_lin, linbuf);

  // CIN (dominant compute): NB batch elements per block
  cin_kernel<<<BATCH / NB, 256, 0, stream>>>(embedB, W0frag, W1cfrag, cinbuf);

  // DNN: h1 = relu(v @ W1 + b1)  -> bf16 (lda 416, pad cols zero-filled)
  gemm_bias_relu<<<dim3(BATCH / 128, 5), 256, 0, stream>>>(
      embedB, FD, 26, W1frag, 25, 5, b1, (void*)h1buf, LDA_H1, 1, 1);
  // DNN: h2 = relu(h1 @ W2 + b2) -> f32
  gemm_bias_relu<<<dim3(BATCH / 128, 5), 256, 0, stream>>>(
      h1buf, LDA_H1, 13, W2frag, 25, 5, b2, (void*)h2buf, U2, 0, 0);

  // final concat-dot + sigmoid
  final_kernel<<<BATCH / 256, 256, 0, stream>>>(linbuf, cinbuf, h2buf, dense,
                                                w_out, b_out, out);
}
